// TriangleMultiplication_47382079209784
// MI455X (gfx1250) — matrix operand order repaired and checked by reading, not compiled, not run on hardware
//
#include <hip/hip_runtime.h>
#include <hip/hip_bf16.h>

// ---------------------------------------------------------------------------
// Triangle multiplication (per-row), L=512, C=128, fp32 in/out.
// All matmuls on v_wmma_f32_16x16x32_bf16 (bf16 operands, f32 accum).
// ---------------------------------------------------------------------------

typedef __attribute__((ext_vector_type(16))) __bf16 v16bf;
typedef __attribute__((ext_vector_type(8)))  float v8f;

union BFrag { v16bf v; unsigned int u[8]; };
union US8   { unsigned short h[8]; int4 v; };

#define L 512
#define C 128

__device__ __forceinline__ unsigned short f2bf(float f) {
    unsigned int u = __float_as_uint(f);
    unsigned int r = u + 0x7FFFu + ((u >> 16) & 1u);   // round-to-nearest-even
    return (unsigned short)(r >> 16);
}
__device__ __forceinline__ float bf2f(unsigned short h) {
    return __uint_as_float(((unsigned int)h) << 16);
}
__device__ __forceinline__ float sigm(float x) {
    return 1.0f / (1.0f + __expf(-x));
}
__device__ __forceinline__ v8f wmma_bf16(const BFrag& a, const BFrag& b, v8f c) {
    return __builtin_amdgcn_wmma_f32_16x16x32_bf16(
        false, a.v, false, b.v, (short)0, c, false, false);
}
__device__ __forceinline__ void load_bfrag(const int4* __restrict__ wp,
                                           int w, int nt, int kc, int lane, BFrag& b) {
    // packed layout: [w][nt][kc][lane][8 dwords]; per-lane 32B == two int4
    const int4* p = wp + (size_t)(((w * 8 + nt) * 4 + kc) * 64 + lane * 2);
    int4 lo = p[0], hi = p[1];
    b.u[0] = lo.x; b.u[1] = lo.y; b.u[2] = lo.z; b.u[3] = lo.w;
    b.u[4] = hi.x; b.u[5] = hi.y; b.u[6] = hi.z; b.u[7] = hi.w;
}

// ---------------------------------------------------------------------------
// Kernel 0: pack 6 fp32 128x128 weights into bf16 WMMA B-fragment layout.
// B 32x16 bf16 layout as used here (contiguous K, matching A): lane%16 = N, lane/16 selects K half (0-15 / 16-31),
// dword d holds K = 2d, 2d+1 (relative).
// ---------------------------------------------------------------------------
__global__ void prepack_weights(const float* __restrict__ w0, const float* __restrict__ w1,
                                const float* __restrict__ w2, const float* __restrict__ w3,
                                const float* __restrict__ w4, const float* __restrict__ w5,
                                unsigned int* __restrict__ wpack) {
    int tid = blockIdx.x * 256 + threadIdx.x;
    if (tid >= 6 * 8192) return;
    int d  = tid & 7;
    int l  = (tid >> 3) & 31;
    int kc = (tid >> 8) & 3;
    int nt = (tid >> 10) & 7;
    int w  = tid >> 13;
    const float* W = (w == 0) ? w0 : (w == 1) ? w1 : (w == 2) ? w2
                   : (w == 3) ? w3 : (w == 4) ? w4 : w5;
    int n = nt * 16 + (l & 15);
    int k = kc * 32 + (l >> 4) * 16 + 2 * d;
    unsigned int lo = f2bf(W[k * C + n]);
    unsigned int hi = f2bf(W[(k + 1) * C + n]);
    wpack[tid] = lo | (hi << 16);
}

// ---------------------------------------------------------------------------
// Kernel 1: fused LayerNorm + 5 projections (left/right/lgate/rgate/fgate).
// One wave per 16-row tile. Writes Lt/Rt as bf16 [c][row][k] (transposed for
// the einsum), gate = sigmoid(x@w_fgate+b) as bf16 [r][c].
// ---------------------------------------------------------------------------
__global__ __launch_bounds__(128)
void fused_ln_proj(const float* __restrict__ act, const float* __restrict__ mask,
                   const float* __restrict__ nsc, const float* __restrict__ nbi,
                   const float* __restrict__ bl,  const float* __restrict__ br,
                   const float* __restrict__ blg, const float* __restrict__ brg,
                   const float* __restrict__ bfg, const int4* __restrict__ wpack,
                   unsigned short* __restrict__ Lt, unsigned short* __restrict__ Rt,
                   unsigned short* __restrict__ gate) {
    __shared__ __align__(16) float          xf[4][16 * C];
    __shared__ __align__(16) unsigned short xh[4][16 * C];

    int wv = threadIdx.x >> 5, lane = threadIdx.x & 31;
    int tile = blockIdx.x * 4 + wv;
    int r0 = tile * 16;
    int i  = r0 >> 9;           // row-pair first index
    int k0 = r0 & (L - 1);      // row-pair second index (16-aligned)
    float*          xfw = xf[wv];
    unsigned short* xhw = xh[wv];

    // --- load 16x128 fp32 tile (fully contiguous 8KB) ---
    const float4* src = (const float4*)(act + (size_t)r0 * C);
    float4* dst = (float4*)xfw;
    #pragma unroll
    for (int t = 0; t < 16; ++t) dst[t * 32 + lane] = src[t * 32 + lane];
    __syncthreads();

    // --- per-row LayerNorm stats: 2 lanes per row ---
    int row = lane >> 1, half = lane & 1;
    float s = 0.f, s2 = 0.f;
    const float* xr = xfw + row * C + half * 64;
    #pragma unroll 8
    for (int e = 0; e < 64; ++e) { float v = xr[e]; s += v; s2 += v * v; }
    s  += __shfl_xor(s, 1);
    s2 += __shfl_xor(s2, 1);
    float mu  = s * (1.0f / C);
    float var = s2 * (1.0f / C) - mu * mu;
    float rs  = rsqrtf(var + 1e-5f);
    #pragma unroll 8
    for (int e = 0; e < 64; ++e) {
        int c = half * 64 + e;
        float v = (xfw[row * C + c] - mu) * rs * nsc[c] + nbi[c];
        xhw[row * C + c] = f2bf(v);
    }
    __syncthreads();

    // --- build 4 A-fragments (16x32 bf16) from LDS via b128 reads ---
    int ll = lane & 15, hl = lane >> 4;
    BFrag A[4];
    #pragma unroll
    for (int kc = 0; kc < 4; ++kc) {
        const int4* p = (const int4*)(xhw + ll * C + kc * 32 + hl * 16);
        int4 lo = p[0], hi = p[1];   // 16 contiguous K per lane half
        A[kc].u[0] = lo.x; A[kc].u[1] = lo.y; A[kc].u[2] = lo.z; A[kc].u[3] = lo.w;
        A[kc].u[4] = hi.x; A[kc].u[5] = hi.y; A[kc].u[6] = hi.z; A[kc].u[7] = hi.w;
    }

    int mbase = hl * 8;
    float mi = mask[i];
    float pm[8];
    #pragma unroll
    for (int d = 0; d < 8; ++d) pm[d] = mi * mask[k0 + mbase + d];

    const v8f vz = {0.f,0.f,0.f,0.f,0.f,0.f,0.f,0.f};
    #pragma unroll 1
    for (int nt = 0; nt < 8; ++nt) {
        int n = nt * 16 + ll;
        BFrag B;
        // ---- left * sigmoid(lgate) ----
        v8f a0 = vz, a1 = vz;
        #pragma unroll
        for (int kc = 0; kc < 4; ++kc) { load_bfrag(wpack, 0, nt, kc, lane, B); a0 = wmma_bf16(A[kc], B, a0); }
        #pragma unroll
        for (int kc = 0; kc < 4; ++kc) { load_bfrag(wpack, 2, nt, kc, lane, B); a1 = wmma_bf16(A[kc], B, a1); }
        {
            float b0 = bl[n], b1 = blg[n];
            US8 o;
            #pragma unroll
            for (int d = 0; d < 8; ++d)
                o.h[d] = f2bf((a0[d] + b0) * pm[d] * sigm(a1[d] + b1));
            *(int4*)(Lt + ((size_t)n * L + i) * L + k0 + mbase) = o.v;
        }
        // ---- right * sigmoid(rgate) ----
        a0 = vz; a1 = vz;
        #pragma unroll
        for (int kc = 0; kc < 4; ++kc) { load_bfrag(wpack, 1, nt, kc, lane, B); a0 = wmma_bf16(A[kc], B, a0); }
        #pragma unroll
        for (int kc = 0; kc < 4; ++kc) { load_bfrag(wpack, 3, nt, kc, lane, B); a1 = wmma_bf16(A[kc], B, a1); }
        {
            float b0 = br[n], b1 = brg[n];
            US8 o;
            #pragma unroll
            for (int d = 0; d < 8; ++d)
                o.h[d] = f2bf((a0[d] + b0) * pm[d] * sigm(a1[d] + b1));
            *(int4*)(Rt + ((size_t)n * L + i) * L + k0 + mbase) = o.v;
        }
        // ---- output gate: sigmoid(x @ w_fgate + b_fgate), natural [r][c] ----
        a0 = vz;
        #pragma unroll
        for (int kc = 0; kc < 4; ++kc) { load_bfrag(wpack, 4, nt, kc, lane, B); a0 = wmma_bf16(A[kc], B, a0); }
        {
            float b0 = bfg[n];
            #pragma unroll
            for (int d = 0; d < 8; ++d)
                gate[(size_t)(r0 + mbase + d) * C + n] = f2bf(sigm(a0[d] + b0));
        }
    }
}

// ---------------------------------------------------------------------------
// Kernel 2: einsum bikc,bjkc->bijc as 128 per-channel 512x512x512 GEMMs.
// Block = (c, 64i x 256j); wave = 16i x 128j with 8 f32 accumulators.
// A/B fragments are contiguous b128 loads from the [c][row][k] bf16 layout.
// Output stored transposed [c][i][j] fp32 for coalesced writes.
// ---------------------------------------------------------------------------
__global__ __launch_bounds__(256)
void einsum_tri(const unsigned short* __restrict__ Lt,
                const unsigned short* __restrict__ Rt,
                float* __restrict__ outT) {
    int wv = threadIdx.x >> 5, lane = threadIdx.x & 31;
    int c  = blockIdx.x & (C - 1);
    int t  = blockIdx.x >> 7;
    int i0 = (t & 7) * 64;
    int j0 = (t >> 3) * 256;
    int iw = i0 + (wv & 3) * 16;
    int jw = j0 + (wv >> 2) * 128;
    int ll = lane & 15, hl = lane >> 4;

    const unsigned short* Lc = Lt + (size_t)c * L * L;
    const unsigned short* Rc = Rt + (size_t)c * L * L;

    v8f acc[8];
    const v8f vz = {0.f,0.f,0.f,0.f,0.f,0.f,0.f,0.f};
    #pragma unroll
    for (int jt = 0; jt < 8; ++jt) acc[jt] = vz;

    #pragma unroll 1
    for (int kc = 0; kc < 16; ++kc) {
        int k0 = kc * 32;
        BFrag A;
        {
            const int4* pa = (const int4*)(Lc + (size_t)(iw + ll) * L + k0 + hl * 16);
            int4 lo = pa[0], hi = pa[1];
            A.u[0] = lo.x; A.u[1] = lo.y; A.u[2] = lo.z; A.u[3] = lo.w;
            A.u[4] = hi.x; A.u[5] = hi.y; A.u[6] = hi.z; A.u[7] = hi.w;
        }
        if (kc + 1 < 16)
            __builtin_prefetch(Lc + (size_t)(iw + ll) * L + k0 + 32, 0, 1);
        #pragma unroll
        for (int jt = 0; jt < 8; ++jt) {
            BFrag B;
            const int4* pb = (const int4*)(Rc + (size_t)(jw + jt * 16 + ll) * L + k0 + hl * 16);
            int4 lo = pb[0], hi = pb[1];
            B.u[0] = lo.x; B.u[1] = lo.y; B.u[2] = lo.z; B.u[3] = lo.w;
            B.u[4] = hi.x; B.u[5] = hi.y; B.u[6] = hi.z; B.u[7] = hi.w;
            acc[jt] = wmma_bf16(A, B, acc[jt]);
        }
    }

    int mb = hl * 8;
    float* oc = outT + (size_t)c * L * L;
    #pragma unroll
    for (int jt = 0; jt < 8; ++jt)
        #pragma unroll
        for (int d = 0; d < 8; ++d)
            oc[(size_t)(iw + mb + d) * L + jw + jt * 16 + ll] = acc[jt][d];
}

// ---------------------------------------------------------------------------
// Kernel 3: fused LayerNorm(fnorm) + out-projection + gate multiply.
// Block = (i, 64 j's). Gathers [c][i][j] tile into LDS (coalesced along j),
// LN over c per (i,j) row, WMMA vs packed w_out, multiply by gate, fp32 out.
// ---------------------------------------------------------------------------
__global__ __launch_bounds__(128)
void fused_fnorm_out(const float* __restrict__ outT,
                     const unsigned short* __restrict__ gate,
                     const float* __restrict__ fsc, const float* __restrict__ fbi,
                     const float* __restrict__ bo,  const int4* __restrict__ wpack,
                     float* __restrict__ out) {
    __shared__ __align__(16) float          xt[64 * 132];   // padded pitch
    __shared__ __align__(16) unsigned short gt[64 * C];

    int i  = blockIdx.x >> 3;
    int j0 = (blockIdx.x & 7) * 64;
    int tid = threadIdx.x;

    // gather 128(c) x 64(j) fp32 tile, coalesced along j
    #pragma unroll 4
    for (int it = 0; it < 64; ++it) {
        int lin = it * 128 + tid;
        int cc = lin >> 6, jl = lin & 63;
        xt[jl * 132 + cc] = outT[(size_t)cc * L * L + (size_t)i * L + j0 + jl];
    }
    // gather gate tile, coalesced along c
    #pragma unroll 4
    for (int it = 0; it < 32; ++it) {
        int lin = it * 128 + tid;
        int jl = lin >> 6;            // 32*128/128 = 32 rows pass 1? -> use 2 elems
        int cc2 = (lin & 63) * 2;
        size_t g = ((size_t)(i * L + j0 + jl)) * C + cc2;
        gt[jl * C + cc2]     = gate[g];
        gt[jl * C + cc2 + 1] = gate[g + 1];
    }
    __syncthreads();

    int wv = tid >> 5, lane = tid & 31;
    int rowb = wv * 16;
    int ll = lane & 15, hl = lane >> 4;

    // LayerNorm over c for 16 rows (2 lanes/row)
    {
        int row = rowb + (lane >> 1), half = lane & 1;
        float s = 0.f, s2 = 0.f;
        #pragma unroll 8
        for (int e = 0; e < 64; ++e) {
            float v = xt[row * 132 + half * 64 + e];
            s += v; s2 += v * v;
        }
        s  += __shfl_xor(s, 1);
        s2 += __shfl_xor(s2, 1);
        float mu  = s * (1.0f / C);
        float var = s2 * (1.0f / C) - mu * mu;
        float rs  = rsqrtf(var + 1e-5f);
        #pragma unroll 8
        for (int e = 0; e < 64; ++e) {
            int cc = half * 64 + e;
            xt[row * 132 + cc] = (xt[row * 132 + cc] - mu) * rs * fsc[cc] + fbi[cc];
        }
    }
    __syncthreads();

    // A fragments from normalized fp32 LDS tile
    BFrag A[4];
    #pragma unroll
    for (int kc = 0; kc < 4; ++kc) {
        const float* base = xt + (rowb + ll) * 132 + kc * 32;
        #pragma unroll
        for (int d = 0; d < 8; ++d) {
            int K = hl * 16 + 2 * d;
            unsigned int lo = f2bf(base[K]);
            unsigned int hi = f2bf(base[K + 1]);
            A[kc].u[d] = lo | (hi << 16);
        }
    }

    const v8f vz = {0.f,0.f,0.f,0.f,0.f,0.f,0.f,0.f};
    #pragma unroll 1
    for (int nt = 0; nt < 8; ++nt) {
        v8f a0 = vz;
        BFrag B;
        #pragma unroll
        for (int kc = 0; kc < 4; ++kc) { load_bfrag(wpack, 5, nt, kc, lane, B); a0 = wmma_bf16(A[kc], B, a0); }
        int n = nt * 16 + ll;
        float bn = bo[n];
        #pragma unroll
        for (int d = 0; d < 8; ++d) {
            int rjl = rowb + hl * 8 + d;
            float g = bf2f(gt[rjl * C + n]);
            out[((size_t)(i * L + j0 + rjl)) * C + n] = (a0[d] + bn) * g;
        }
    }
}

// ---------------------------------------------------------------------------
// launch
// ---------------------------------------------------------------------------
extern "C" void kernel_launch(void* const* d_in, const int* in_sizes, int n_in,
                              void* d_out, int out_size, void* d_ws, size_t ws_size,
                              hipStream_t stream) {
    const float* act  = (const float*)d_in[0];
    const float* mask = (const float*)d_in[1];
    const float* nsc  = (const float*)d_in[2];
    const float* nbi  = (const float*)d_in[3];
    const float* w_l  = (const float*)d_in[4];
    const float* b_l  = (const float*)d_in[5];
    const float* w_r  = (const float*)d_in[6];
    const float* b_r  = (const float*)d_in[7];
    const float* w_lg = (const float*)d_in[8];
    const float* b_lg = (const float*)d_in[9];
    const float* w_rg = (const float*)d_in[10];
    const float* b_rg = (const float*)d_in[11];
    const float* fsc  = (const float*)d_in[12];
    const float* fbi  = (const float*)d_in[13];
    const float* w_o  = (const float*)d_in[14];
    const float* b_o  = (const float*)d_in[15];
    const float* w_fg = (const float*)d_in[16];
    const float* b_fg = (const float*)d_in[17];
    float* out = (float*)d_out;

    char* ws = (char*)d_ws;
    const size_t HALF = (size_t)C * L * L * sizeof(unsigned short); // 64 MiB
    unsigned int*   wpack = (unsigned int*)ws;                      // 192 KiB
    unsigned short* Lt    = (unsigned short*)(ws + (size_t)262144);
    unsigned short* Rt    = (unsigned short*)(ws + (size_t)262144 + HALF);
    unsigned short* gateb = (unsigned short*)(ws + (size_t)262144 + 2 * HALF);
    float*          outT  = (float*)(ws + (size_t)262144 + 3 * HALF);

    prepack_weights<<<192, 256, 0, stream>>>(w_l, w_r, w_lg, w_rg, w_fg, w_o, wpack);

    fused_ln_proj<<<4096, 128, 0, stream>>>(act, mask, nsc, nbi,
                                            b_l, b_r, b_lg, b_rg, b_fg,
                                            (const int4*)wpack, Lt, Rt, gateb);

    einsum_tri<<<2048, 256, 0, stream>>>(Lt, Rt, outT);

    fused_fnorm_out<<<4096, 128, 0, stream>>>(outT, gateb, fsc, fbi, b_o,
                                              (const int4*)wpack, out);
}
